// GRU_44813688767109
// MI455X (gfx1250) — compile-verified, matrix operand order checked by reading, not run on hardware
//
#include <hip/hip_runtime.h>

// ---------------------------------------------------------------------------
// 2-layer GRU for MI455X (gfx1250, wave32, WMMA).
//   Phase A (per layer): gates_x = X @ W_ih^T + b_ih  -> big bf16 WMMA GEMM
//   Phase B (per layer): persistent scan kernel, grid barrier per timestep.
//     W_hh slice (96KB) pre-swizzled into LDS once via async global->LDS;
//     per step only h (A operand) comes from L2.
// ---------------------------------------------------------------------------

typedef __attribute__((ext_vector_type(16))) __bf16 v16bf;
typedef __attribute__((ext_vector_type(8)))  __bf16 v8bf;
typedef __attribute__((ext_vector_type(8)))  float  v8f;
typedef int v4i __attribute__((vector_size(16)));   // builtin's pointee type
typedef __attribute__((address_space(3))) v4i* lds_v4i;

#define T_STEPS 512
#define B_SZ    64
#define I_DIM   1024
#define HD      1024
#define G3      (3 * HD)
#define NLAYERS 2

// Dynamic LDS layout for gru_scan:
//   [0, 96KB)        : 96 B-fragments (3 gates x 32 ksteps), 512 bf16 each,
//                      pre-swizzled lane-major (lane*32B, two 16B chunks)
//   [96KB, 96KB+12KB): gh exchange [4 Mtile][3 gate][16][16] f32
#define BFRAGS      96
#define SMEM_BYTES  (BFRAGS * 1024 + 4 * 3 * 16 * 16 * 4)

#if defined(__HIP_DEVICE_COMPILE__) && \
    __has_builtin(__builtin_amdgcn_global_load_async_to_lds_b128)
#define USE_ASYNC_LDS 1
#else
#define USE_ASYNC_LDS 0
#endif

// --- WMMA bf16 fragment loader (global/generic, row-major, leading dim ld) ---
// ISA layout for 16-bit A/B operands: lane L holds row (L&15); lanes 0-15 get
// K = [0..8)+[16..24), lanes 16-31 get K = [8..16)+[24..32)  -> two b128 loads.
__device__ __forceinline__ v16bf load_frag(const __bf16* __restrict__ base, int ld) {
  const int lane = threadIdx.x & 31;
  const int row  = lane & 15;
  const int kb   = (lane >> 4) << 3;
  const __bf16* p = base + (size_t)row * ld + kb;
  v8bf lo = *(const v8bf*)(p);
  v8bf hi = *(const v8bf*)(p + 16);
  v16bf f;
#pragma unroll
  for (int i = 0; i < 8; ++i) { f[i] = lo[i]; f[i + 8] = hi[i]; }
  return f;
}

// --- pre-swizzled LDS fragment loader: lane's 32B are contiguous -------------
__device__ __forceinline__ v16bf load_frag_lds(const __bf16* frag) {
  const int lane = threadIdx.x & 31;
  const __bf16* p = frag + lane * 16;
  v8bf lo = *(const v8bf*)(p);
  v8bf hi = *(const v8bf*)(p + 8);
  v16bf f;
#pragma unroll
  for (int i = 0; i < 8; ++i) { f[i] = lo[i]; f[i + 8] = hi[i]; }
  return f;
}

__device__ __forceinline__ v8f wmma_bf16(v16bf a, v16bf b, v8f c) {
  return __builtin_amdgcn_wmma_f32_16x16x32_bf16(false, a, false, b,
                                                 (short)0, c, false, false);
}

__device__ __forceinline__ float sigmoidf_fast(float x) {
  return 1.0f / (1.0f + __expf(-x));
}

// --- fp32 -> bf16 conversion --------------------------------------------------
__global__ void f32_to_bf16_k(const float* __restrict__ src,
                              __bf16* __restrict__ dst, int n) {
  int i = blockIdx.x * blockDim.x + threadIdx.x;
  if (i < n) dst[i] = (__bf16)src[i];
}

// --- h0 init: keep both f32 and bf16 copies ----------------------------------
__global__ void h_init_k(const float* __restrict__ h0,
                         __bf16* __restrict__ hbf, float* __restrict__ hf, int n) {
  int i = blockIdx.x * blockDim.x + threadIdx.x;
  if (i < n) { float v = h0[i]; hf[i] = v; hbf[i] = (__bf16)v; }
}

// --- Phase A: gates_x GEMM ----------------------------------------------------
// C[M=32768, N=3072] = A[M,K=1024](bf16) x W^T (W is [N,K] row-major, bf16) + bias
// Block tile 128x128, 8 waves as 4x2, wave tile 32x64 (2x4 WMMA tiles).
__global__ __launch_bounds__(256) void gemm_gates_x(
    const __bf16* __restrict__ A, const __bf16* __restrict__ W,
    const float* __restrict__ bias, float* __restrict__ C) {
  const int K = I_DIM;
  const int wave = threadIdx.x >> 5;
  const int wm = wave & 3, wn = wave >> 2;
  const int m0 = blockIdx.x * 128 + wm * 32;
  const int n0 = blockIdx.y * 128 + wn * 64;

  v8f acc[2][4];
#pragma unroll
  for (int i = 0; i < 2; ++i)
#pragma unroll
    for (int j = 0; j < 4; ++j) acc[i][j] = (v8f){};

#pragma unroll 2
  for (int k0 = 0; k0 < K; k0 += 32) {
    v16bf a[2], b[4];
#pragma unroll
    for (int i = 0; i < 2; ++i)
      a[i] = load_frag(A + (size_t)(m0 + 16 * i) * K + k0, K);
#pragma unroll
    for (int j = 0; j < 4; ++j)
      b[j] = load_frag(W + (size_t)(n0 + 16 * j) * K + k0, K);
#pragma unroll
    for (int i = 0; i < 2; ++i)
#pragma unroll
      for (int j = 0; j < 4; ++j) acc[i][j] = wmma_bf16(a[i], b[j], acc[i][j]);
  }

  // C/D layout: VGPR v -> row v + 8*(lane>>4); col = lane&15
  const int lane = threadIdx.x & 31;
  const int col = lane & 15, rb = (lane >> 4) << 3;
#pragma unroll
  for (int i = 0; i < 2; ++i)
#pragma unroll
    for (int j = 0; j < 4; ++j)
#pragma unroll
      for (int v = 0; v < 8; ++v) {
        int r = m0 + 16 * i + rb + v;
        int c = n0 + 16 * j + col;
        C[(size_t)r * G3 + c] = acc[i][j][v] + bias[c];
      }
}

// --- Phase B: persistent recurrent scan --------------------------------------
// 64 blocks x 128 threads. Block jc owns hidden columns [16*jc, 16*jc+16) and
// computes all three gates for them. Wave m owns batch-rows [16m, 16m+16).
// B operand (W_hh slice) lives in LDS for the whole scan; only h streams in.
__global__ __launch_bounds__(128) void gru_scan(
    const float* __restrict__ gx,      // [T, B, 3H] (b_ih included)
    const __bf16* __restrict__ Whh,    // [3H, H] bf16 (L2-resident)
    const float* __restrict__ bhh,     // [3H]
    __bf16* __restrict__ hbf0, __bf16* __restrict__ hbf1,
    float* __restrict__ hf0, float* __restrict__ hf1,
    __bf16* __restrict__ ybf,          // next-layer input (bf16) or nullptr
    float* __restrict__ yf32,          // final output (f32) or nullptr
    float* __restrict__ hn_out,        // d_out slice for final h of this layer
    unsigned* __restrict__ bar) {
  extern __shared__ __align__(16) char smem_raw[];
  __bf16* Bsh  = (__bf16*)smem_raw;                   // 96 frags x 512 bf16
  float*  ghsh = (float*)(smem_raw + BFRAGS * 1024);  // [4][3][16][16]

  const int jc   = blockIdx.x;         // column tile 0..63
  const int wave = threadIdx.x >> 5;   // M tile 0..3
  const int lane = threadIdx.x & 31;
  const unsigned nb = gridDim.x;

  // ---- one-time preload: swizzle W_hh slice into fragment-lane order -------
  // frag f = g*32 + ks holds B tile (cols 16*jc..+16, k = 32*ks..+32);
  // lane's 32B: [k=kb..kb+8) then [k=kb+16..kb+24), kb = (lane>>4)*8.
  {
    const int row = lane & 15;
    const int kb  = (lane >> 4) << 3;
    for (int f = wave; f < BFRAGS; f += 4) {
      const int g = f >> 5, ks = f & 31;
      const __bf16* src =
          Whh + (size_t)(g * HD + jc * 16 + row) * HD + ks * 32 + kb;
      __bf16* dst = Bsh + f * 512 + lane * 16;
#if USE_ASYNC_LDS
      __builtin_amdgcn_global_load_async_to_lds_b128(
          (v4i*)src, (lds_v4i)dst, 0, 0);
      __builtin_amdgcn_global_load_async_to_lds_b128(
          (v4i*)(src + 16), (lds_v4i)(dst + 8), 0, 0);
#else
      *(v8bf*)dst       = *(const v8bf*)src;        // low K chunk
      *(v8bf*)(dst + 8) = *(const v8bf*)(src + 16); // high K chunk
#endif
    }
#if USE_ASYNC_LDS
#if __has_builtin(__builtin_amdgcn_s_wait_asynccnt)
    __builtin_amdgcn_s_wait_asynccnt(0);
#else
    asm volatile("s_wait_asynccnt 0" ::: "memory");
#endif
#endif
  }
  __syncthreads();

  for (int t = 0; t < T_STEPS; ++t) {
    const __bf16* hprev  = (t & 1) ? hbf1 : hbf0;
    const float*  hfprev = (t & 1) ? hf1  : hf0;
    __bf16* hcur  = (t & 1) ? hbf0 : hbf1;
    float*  hfcur = (t & 1) ? hf0  : hf1;

    v8f acc[3];
#pragma unroll
    for (int g = 0; g < 3; ++g) acc[g] = (v8f){};

    const __bf16* abase = hprev + (size_t)(wave * 16) * HD;
#pragma unroll 2
    for (int ks = 0; ks < 32; ++ks) {
      v16bf a = load_frag(abase + ks * 32, HD);
#pragma unroll
      for (int g = 0; g < 3; ++g) {
        v16bf b = load_frag_lds(Bsh + (g * 32 + ks) * 512);
        acc[g] = wmma_bf16(a, b, acc[g]);
      }
    }

    // park gh tiles in LDS for the fused gate math
    const int col = lane & 15, rb = (lane >> 4) << 3;
#pragma unroll
    for (int g = 0; g < 3; ++g)
#pragma unroll
      for (int v = 0; v < 8; ++v)
        ghsh[((wave * 3 + g) * 16 + rb + v) * 16 + col] = acc[g][v];
    __syncthreads();

    // fused elementwise GRU update: 64 rows x 16 cols = 1024 elements
    const size_t tb = (size_t)t * B_SZ;
    for (int e = threadIdx.x; e < B_SZ * 16; e += blockDim.x) {
      const int row = e >> 4;
      const int c16 = e & 15;
      const int ch  = jc * 16 + c16;
      const float* gxp = gx + (tb + row) * (size_t)G3;
      const int mt = row >> 4, rl = row & 15;

      float gr = gxp[ch]          + ghsh[((mt * 3 + 0) * 16 + rl) * 16 + c16] + bhh[ch];
      float gz = gxp[HD + ch]     + ghsh[((mt * 3 + 1) * 16 + rl) * 16 + c16] + bhh[HD + ch];
      float gn = gxp[2 * HD + ch];
      float hg = ghsh[((mt * 3 + 2) * 16 + rl) * 16 + c16] + bhh[2 * HD + ch];

      float r = sigmoidf_fast(gr);
      float z = sigmoidf_fast(gz);
      float n = tanhf(gn + r * hg);
      float hold = hfprev[row * HD + ch];
      float hnew = (1.0f - z) * n + z * hold;

      hfcur[row * HD + ch] = hnew;
      hcur[row * HD + ch]  = (__bf16)hnew;
      if (ybf)  ybf[(tb + row) * (size_t)HD + ch]  = (__bf16)hnew;
      if (yf32) yf32[(tb + row) * (size_t)HD + ch] = hnew;
      if (t == T_STEPS - 1) hn_out[row * HD + ch] = hnew;
    }

    // grid-wide barrier: monotonic counter, no reset race
    __threadfence();
    __syncthreads();
    if (threadIdx.x == 0) {
      __hip_atomic_fetch_add(bar, 1u, __ATOMIC_ACQ_REL, __HIP_MEMORY_SCOPE_AGENT);
      const unsigned target = (unsigned)(t + 1) * nb;
      while (__hip_atomic_load(bar, __ATOMIC_ACQUIRE, __HIP_MEMORY_SCOPE_AGENT) <
             target)
        __builtin_amdgcn_s_sleep(1);
    }
    __syncthreads();
  }
}

// ---------------------------------------------------------------------------
extern "C" void kernel_launch(void* const* d_in, const int* in_sizes, int n_in,
                              void* d_out, int out_size, void* d_ws,
                              size_t ws_size, hipStream_t stream) {
  const float* x   = (const float*)d_in[0];  // [T,B,I]
  const float* H0  = (const float*)d_in[1];  // [L,B,H]
  const float* Wih = (const float*)d_in[2];  // [L,3H,I]
  const float* Whh = (const float*)d_in[3];  // [L,3H,H]
  const float* bih = (const float*)d_in[4];  // [L,3H]
  const float* bhh = (const float*)d_in[5];  // [L,3H]
  float* out = (float*)d_out;                // [T,B,H] ++ [L,B,H]

  char* ws = (char*)d_ws;
  size_t off = 0;
  auto alloc = [&](size_t bytes) -> void* {
    void* p = ws + off;
    off = (off + bytes + 255) & ~(size_t)255;
    return p;
  };
  __bf16* xb   = (__bf16*)alloc((size_t)T_STEPS * B_SZ * I_DIM * 2);
  float*  gx   = (float*) alloc((size_t)T_STEPS * B_SZ * G3 * 4);
  __bf16* wihb = (__bf16*)alloc((size_t)G3 * I_DIM * 2);
  __bf16* whhb = (__bf16*)alloc((size_t)G3 * HD * 2);
  __bf16* hbf0 = (__bf16*)alloc((size_t)B_SZ * HD * 2);
  __bf16* hbf1 = (__bf16*)alloc((size_t)B_SZ * HD * 2);
  float*  hf0  = (float*) alloc((size_t)B_SZ * HD * 4);
  float*  hf1  = (float*) alloc((size_t)B_SZ * HD * 4);
  unsigned* bar = (unsigned*)alloc(256);

  // allow >64KB dynamic LDS for the scan kernel (CDNA5 WGP has 320KB)
  (void)hipFuncSetAttribute((const void*)gru_scan,
                            hipFuncAttributeMaxDynamicSharedMemorySize,
                            SMEM_BYTES);

  const int nw  = G3 * I_DIM;                 // weight elems per matrix
  const int nx  = T_STEPS * B_SZ * I_DIM;     // input elems
  const int nh  = B_SZ * HD;
  const dim3 gemm_grid((T_STEPS * B_SZ) / 128, G3 / 128);  // 256 x 24

  for (int l = 0; l < NLAYERS; ++l) {
    f32_to_bf16_k<<<(nw + 255) / 256, 256, 0, stream>>>(
        Wih + (size_t)l * nw, wihb, nw);
    f32_to_bf16_k<<<(nw + 255) / 256, 256, 0, stream>>>(
        Whh + (size_t)l * nw, whhb, nw);
    if (l == 0)
      f32_to_bf16_k<<<(nx + 255) / 256, 256, 0, stream>>>(x, xb, nx);
    // (for l==1, xb already holds layer-0 output written by gru_scan)

    gemm_gates_x<<<gemm_grid, 256, 0, stream>>>(xb, wihb,
                                                bih + (size_t)l * G3, gx);

    h_init_k<<<(nh + 255) / 256, 256, 0, stream>>>(H0 + (size_t)l * nh, hbf0,
                                                   hf0, nh);
    (void)hipMemsetAsync(bar, 0, sizeof(unsigned), stream);

    gru_scan<<<64, 128, SMEM_BYTES, stream>>>(
        gx, whhb, bhh + (size_t)l * G3, hbf0, hbf1, hf0, hf1,
        (l == 0) ? xb : nullptr,                       // next-layer input
        (l == NLAYERS - 1) ? out : nullptr,            // final f32 output
        out + (size_t)T_STEPS * B_SZ * HD + (size_t)l * B_SZ * HD, bar);
  }
}
